// RNNJANSEN_83124797047477
// MI455X (gfx1250) — matrix operand order repaired and checked by reading, not compiled, not run on hardware
//
#include <hip/hip_runtime.h>
#include <hip/hip_bf16.h>

typedef __attribute__((ext_vector_type(16))) _Float16 v16h;
typedef __attribute__((ext_vector_type(8)))  float    v8f;

#define NODE   200
#define HIDDEN 10
#define BATCH  50
#define NOUT   64
#define BUF    500
#define HISTP  512   // padded circular history depth (power of 2)
#define KPAD   224   // 200 padded up to multiple of 32 for WMMA K

// d_out layout (floats, reference return order):
//   eeg      [64*50]   @ 0
//   state    [200*6]   @ 3200
//   Eb.T     [200*50]  @ 4400
//   Ib.T     [200*50]  @ 14400
//   Mb.T     [200*50]  @ 24400
//   Evb.T    [200*50]  @ 34400
//   Ivb.T    [200*50]  @ 44400
//   Mvb.T    [200*50]  @ 54400
//   hEb      [200*500] @ 64400
#define OFF_EEG  0
#define OFF_CS   3200
#define OFF_EB   4400
#define OFF_IB   14400
#define OFF_MB   24400
#define OFF_EVB  34400
#define OFF_IVB  44400
#define OFF_MVB  54400
#define OFF_HEB  64400

__launch_bounds__(256)
__global__ void rnnjansen_persistent(
    const float* __restrict__ inp, const float* __restrict__ noise_in,
    const float* __restrict__ hx,  const float* __restrict__ hE,
    const float* __restrict__ sc,  const float* __restrict__ w_bb,
    const float* __restrict__ lm,  const int*   __restrict__ delays,
    float* __restrict__ out, float* __restrict__ ws)
{
  const int tid = threadIdx.x;

  // workspace (floats): wn[200*200] | hist[200*512] | lmn[64*200]
  float* wn   = ws;
  float* hist = ws + 40000;
  float* lmn  = ws + 40000 + NODE*HISTP;

  __shared__ float    red[256];
  __shared__ float    sMean[KPAD];
  __shared__ _Float16 Astage[NOUT][KPAD];   // lm_t as f16, K-padded
  __shared__ _Float16 Bstage[KPAD][NOUT];   // (Eb - Ib) as f16, K x N

  // Compile-time folded parameters (relu of positive constants)
  const float DT   = 1e-4f;
  const float A_e  = 3.25f,    a_e = 101.0f;
  const float B_e  = 22.0f,    b_e = 51.0f;
  const float g_e  = 1000.01f, c1e = 135.01f, c2e = 108.01f;
  const float c3e  = 33.76f,   c4e = 33.76f;
  const float std_e = 250.0f;
  const float U_UBc = 500.0f,  SATc = 1000.0f;
  const float vmax = 5.0f, v0 = 6.0f, r_ = 0.56f;
  const float cy0 = 5.0f, y0 = 2.0f;

  // ---------- Phase 0: wl = log1p(0.5*(w + w^T)), Frobenius norm ----------
  float partial = 0.f;
  for (int idx = tid; idx < NODE*NODE; idx += 256) {
    int i = idx / NODE, j = idx % NODE;
    float wij = expf(w_bb[i*NODE+j]) * sc[i*NODE+j];
    float wji = expf(w_bb[j*NODE+i]) * sc[j*NODE+i];
    float wl  = log1pf(0.5f*(wij + wji));
    wn[idx] = wl;
    partial += wl*wl;
  }
  red[tid] = partial;
  __syncthreads();
  for (int s = 128; s > 0; s >>= 1) {
    if (tid < s) red[tid] += red[tid + s];
    __syncthreads();
  }
  const float inv_norm = 1.0f / sqrtf(red[0]);
  __threadfence();           // wl writes visible before scaling pass
  __syncthreads();

  // scale rows in place, compute per-node row-sum (wn is symmetric)
  float rs = 0.f;
  if (tid < NODE) {
    for (int j = 0; j < NODE; ++j) {
      float v = wn[tid*NODE + j] * inv_norm;
      wn[tid*NODE + j] = v;
      rs += v;
    }
  }

  // ---------- Phase 1: state in registers, history circular buffer ----------
  float M=0.f, E=0.f, I=0.f, Mv=0.f, Ev=0.f, Iv=0.f;
  if (tid < NODE) {
    M  = hx[tid*6+0]; E  = hx[tid*6+1]; I  = hx[tid*6+2];
    Mv = hx[tid*6+3]; Ev = hx[tid*6+4]; Iv = hx[tid*6+5];
  }
  for (int idx = tid; idx < NODE*HISTP; idx += 256) {
    int n = idx / HISTP, k = idx % HISTP;
    hist[idx] = (k < BUF) ? hE[n*BUF + k] : 0.f;
  }
  __threadfence();
  __syncthreads();

  // ---------- Phase 2: 500 sequential Euler steps ----------
  int head = 0;  // logical delay d -> physical (head + d) & 511
  for (int t = 0; t < 500; ++t) {
    const int b = t / HIDDEN, h = t % HIDDEN;
    if (tid < NODE) {
      const int i = tid;
      // delayed coupling: LEd[i] = sum_j wn[i,j] * hist[j, delays[j,i]]
      float acc = 0.f;
      for (int j = 0; j < NODE; ++j) {
        int d = delays[j*NODE + i];
        acc = fmaf(wn[j*NODE + i],                       // symmetric: coalesced
                   hist[j*HISTP + ((head + d) & (HISTP-1))], acc);
      }
      const float u  = inp[i*(HIDDEN*BATCH) + h*BATCH + b];
      const float nz = noise_in[i*(HIDDEN*BATCH) + h*BATCH + b];

      float rM = vmax / (1.f + expf(r_*(v0 - (E - I))));
      float rE = std_e*nz + g_e*(acc - rs*E)
               + c2e*(vmax / (1.f + expf(r_*(v0 - c1e*M))));
      float rI = c4e*(vmax / (1.f + expf(r_*(v0 - c3e*M))));

      float ddM = M + DT*Mv;
      float ddE = E + DT*Ev;
      float ddI = I + DT*Iv;
      float uM = u + U_UBc*tanhf(rM/U_UBc);
      float uE =     U_UBc*tanhf(rE/U_UBc);
      float uI =     U_UBc*tanhf(rI/U_UBc);
      float ddMv = Mv + DT*(A_e*a_e*uM - 2.f*a_e*Mv - a_e*a_e*M);
      float ddEv = Ev + DT*(A_e*a_e*uE - 2.f*a_e*Ev - a_e*a_e*E);
      float ddIv = Iv + DT*(B_e*b_e*uI - 2.f*b_e*Iv - b_e*b_e*I);

      M  = SATc*tanhf(ddM /SATc);
      E  = SATc*tanhf(ddE /SATc);
      I  = SATc*tanhf(ddI /SATc);
      Mv = SATc*tanhf(ddMv/SATc);
      Ev = SATc*tanhf(ddEv/SATc);
      Iv = SATc*tanhf(ddIv/SATc);

      // push new M: write slot head-1 (outside this step's read window)
      hist[i*HISTP + ((head - 1) & (HISTP-1))] = M;

      if (h == HIDDEN-1) {   // record end-of-trial state (recs)
        out[OFF_EB  + i*BATCH + b] = E;
        out[OFF_IB  + i*BATCH + b] = I;
        out[OFF_MB  + i*BATCH + b] = M;
        out[OFF_EVB + i*BATCH + b] = Ev;
        out[OFF_IVB + i*BATCH + b] = Iv;
        out[OFF_MVB + i*BATCH + b] = Mv;
      }
    }
    head = (head - 1) & (HISTP-1);
    __threadfence();
    __syncthreads();
  }

  // ---------- Phase 3: current_state and hEb outputs ----------
  if (tid < NODE) {
    out[OFF_CS + tid*6+0] = M;  out[OFF_CS + tid*6+1] = E;
    out[OFF_CS + tid*6+2] = I;  out[OFF_CS + tid*6+3] = Mv;
    out[OFF_CS + tid*6+4] = Ev; out[OFF_CS + tid*6+5] = Iv;
  }
  for (int idx = tid; idx < NODE*BUF; idx += 256) {
    int n = idx / BUF, d = idx % BUF;
    out[OFF_HEB + idx] = hist[n*HISTP + ((head + d) & (HISTP-1))];
  }

  // ---------- Phase 4: eeg = cy0 * lm_t @ (Eb - Ib) - y0 via WMMA ----------
  // lm row-normalize
  if (tid < NOUT) {
    float ss = 0.f;
    for (int j = 0; j < NODE; ++j) { float v = lm[tid*NODE+j]; ss += v*v; }
    float inv = 1.f / sqrtf(ss);
    for (int j = 0; j < NODE; ++j) lmn[tid*NODE+j] = lm[tid*NODE+j]*inv;
  }
  __threadfence();
  __syncthreads();
  // column means over the 64 output rows
  if (tid < KPAD) {
    float m = 0.f;
    if (tid < NODE) {
      for (int o = 0; o < NOUT; ++o) m += lmn[o*NODE + tid];
      m *= (1.0f/NOUT);
    }
    sMean[tid] = m;
  }
  __syncthreads();
  // stage A = lm_t (64 x 224, f16) and B = Eb - Ib (224 x 64, f16)
  for (int idx = tid; idx < NOUT*KPAD; idx += 256) {
    int o = idx / KPAD, k = idx % KPAD;
    float v = (k < NODE) ? (lmn[o*NODE + k] - sMean[k]) : 0.f;
    Astage[o][k] = (_Float16)v;
  }
  for (int idx = tid; idx < KPAD*NOUT; idx += 256) {
    int k = idx / NOUT, bb = idx % NOUT;
    float v = 0.f;
    if (k < NODE && bb < BATCH)
      v = out[OFF_EB + k*BATCH + bb] - out[OFF_IB + k*BATCH + bb];
    Bstage[k][bb] = (_Float16)v;
  }
  __syncthreads();

  // 4x4 grid of 16x16 output tiles over 8 waves (2 tiles/wave), K = 7 x 32
  const int lane = tid & 31;
  const int wave = tid >> 5;
  const int lrow = lane & 15;
  const int hi   = lane >> 4;            // lane half select
  for (int tile = wave; tile < 16; tile += 8) {
    const int mbase = (tile >> 2) * 16;  // output-row base
    const int nbase = (tile & 3) * 16;   // batch-col base
    v8f acc = {};
    for (int kt = 0; kt < KPAD/32; ++kt) {
      v16h afr, bfr;
      // A fragment: M = mbase+lrow; K = [0..15] (lanes<16) / [16..31] (lanes>=16)
      const int koffA = hi * 8;
      #pragma unroll
      for (int e = 0; e < 16; ++e) {
        int kk = kt*32 + e + koffA * 2;
        afr[e] = Astage[mbase + lrow][kk];
      }
      // B fragment: N = nbase+lrow; K = [0..15] (lanes<16) / [16..31] (lanes>=16)
      const int koffB = hi * 16;
      #pragma unroll
      for (int e = 0; e < 16; ++e) {
        bfr[e] = Bstage[kt*32 + koffB + e][nbase + lrow];
      }
      acc = __builtin_amdgcn_wmma_f32_16x16x32_f16(
          /*neg_a=*/false, afr, /*neg_b=*/false, bfr,
          /*c_mod=*/(short)0, acc, /*reuse_a=*/false, /*reuse_b=*/false);
    }
    // D layout: lanes 0-15 -> M=r, lanes 16-31 -> M=r+8; N = lane%16
    const int col = nbase + lrow;
    #pragma unroll
    for (int r = 0; r < 8; ++r) {
      int o = mbase + r + hi*8;
      if (col < BATCH) out[OFF_EEG + o*BATCH + col] = cy0*acc[r] - y0;
    }
  }
}

extern "C" void kernel_launch(void* const* d_in, const int* in_sizes, int n_in,
                              void* d_out, int out_size, void* d_ws, size_t ws_size,
                              hipStream_t stream) {
  (void)in_sizes; (void)n_in; (void)out_size; (void)ws_size;
  rnnjansen_persistent<<<dim3(1), dim3(256), 0, stream>>>(
      (const float*)d_in[0],  // inp
      (const float*)d_in[1],  // noise_in
      (const float*)d_in[2],  // hx
      (const float*)d_in[3],  // hE
      (const float*)d_in[4],  // sc
      (const float*)d_in[5],  // w_bb
      (const float*)d_in[6],  // lm
      (const int*)d_in[7],    // delays
      (float*)d_out, (float*)d_ws);
}
